// RealNVP_2027224564341
// MI455X (gfx1250) — matrix operand order repaired and checked by reading, compiled, not run on hardware
//
#include <hip/hip_runtime.h>

// ---------------------------------------------------------------------------
// Fused RealNVP inverse pass for MI455X (gfx1250, wave32).
//  - all 6 coupling layers fused; each wave keeps a 16x128 row tile in VGPRs
//  - all GEMMs via v_wmma_f32_16x16x32_bf16 (fp32 accumulate)
//  - weights pre-swizzled once per block into bf16 B-fragment order in LDS
//  - C-layout -> A-layout transitions via tiny per-wave LDS bounce buffers
//  - hardware TRANS ops (v_tanh_f32 / v_exp_f32) for the coupling nonlinearity
//  - non-temporal x/y streaming + next-tile global prefetch
//  - no VGPR clamp: 2-blocks/WGP tier needs <=256 VGPRs which provably spills;
//    spill-free code at 1 block/WGP (2 waves/SIMD) is the better tradeoff
// ---------------------------------------------------------------------------

typedef __attribute__((ext_vector_type(16))) __bf16 v16bf;
typedef __attribute__((ext_vector_type(8)))  float  v8f;
typedef __attribute__((ext_vector_type(4)))  float  v4f;

#define LAYERS 6
#define WPB    8            // waves per block
#define NTH    (WPB * 32)   // 256 threads

// LDS layout (bytes)
#define NET_STRIDE 4864                    // bf16 elems per (layer, net)
#define OFF_BIA    116736                  // 12*4864*2 bytes of weight frags
#define OFF_MSK    (OFF_BIA + 2304 * 4)    // 12*192 f32 biases
#define OFF_SCR    (OFF_MSK + 768 * 4)     // 6*128 f32 masks
#define OFF_ZERO   (OFF_SCR + WPB * 768 * 4)  // 64B zero page for K>=16 lanes
#define SMEM_BYTES (OFF_ZERO + 64)

struct Params {
    const float* x;            // [B,128]
    const float* masks;        // [6,128]
    const float* W[2][5];      // [p][j]: (6, din, dout) row-major
    const float* b[2][5];      // [p][j]: (6, dout)
    float* y;                  // [B,128]
    float* ld;                 // [B]
    int ntiles;                // B/16
};

// exp(x) via hardware v_exp_f32 (computes 2^x)
static __device__ __forceinline__ float fast_exp(float x) {
    return __builtin_amdgcn_exp2f(x * 1.4426950408889634f);
}
// tanh via hardware v_tanh_f32 (confirmed lowering on this toolchain)
static __device__ __forceinline__ float fast_tanh(float x) {
#if __has_builtin(__builtin_amdgcn_tanhf)
    return __builtin_amdgcn_tanhf(x);
#else
    float e = __builtin_amdgcn_exp2f(x * 2.8853900817779268f); // 2*log2(e)*x
    return 1.0f - 2.0f * __builtin_amdgcn_rcpf(e + 1.0f);
#endif
}

// C-tile (v8f, N=laneM across lanes, M in vgprs) -> bias + relu -> bf16
// A-fragment for the next 16x16x32 WMMA (upper K half zero).
static __device__ __forceinline__
v16bf c_to_afrag(v8f acc, const float* brow, float* hbuf, int laneM, int laneHi) {
    const float bN = brow[laneM];
#pragma unroll
    for (int r = 0; r < 8; r++) {
        float v = fmaxf(acc[r] + bN, 0.0f);
        hbuf[(r + 8 * laneHi) * 16 + laneM] = v;      // row-major h[M][N]
    }
    v16bf Ah;
    const float* src = hbuf + laneM * 16; // A layout: M=laneM, K runs
#pragma unroll
    for (int e = 0; e < 16; e++) Ah[e] = laneHi ? (__bf16)0.0f : (__bf16)src[e];   // K = 16*laneHi + e; K >= 16 padded zero
    return Ah;
}

__global__ __launch_bounds__(NTH)
void realnvp_fused(Params P) {
    extern __shared__ __align__(128) char smem[];
    __bf16* wfrag   = (__bf16*)smem;
    float*  bia     = (float*)(smem + OFF_BIA);
    float*  msk     = (float*)(smem + OFF_MSK);
    float*  scratch = (float*)(smem + OFF_SCR);
    const __bf16* zpage = (const __bf16*)(smem + OFF_ZERO);

    const int tid = threadIdx.x;

    // ---- one-time per block: swizzle weights into WMMA B-fragment order ----
    // W1 (128x16): 4 K-chunks, full 32-lane fragments. frag elem (lane,e):
//   K = 32c + 16*(e>=8) + 8*(lane>=16) + (e&7), N = lane&15
    for (int idx = tid; idx < 12 * 2048; idx += NTH) {
        int ln = idx >> 11, r = idx & 2047;
        int l = ln >> 1, p = ln & 1;
        int c = r >> 9, le = r & 511;
        int lane = le >> 4, e = le & 15;
        int K = 32 * c + ((e >> 3) << 4) + ((lane >> 4) << 3) + (e & 7);
        int N = lane & 15;
        wfrag[ln * NET_STRIDE + r] = (__bf16)P.W[p][0][(l * 128 + K) * 16 + N];
    }
    // W2..W4 (16x16): compact fragments, lanes 0..15 only (K = e, N = lane)
    for (int idx = tid; idx < 12 * 768; idx += NTH) {
        int ln = idx / 768, r = idx - ln * 768;
        int l = ln >> 1, p = ln & 1;
        int j = r >> 8, le = r & 255;
        int lane = le >> 4, e = le & 15;
        wfrag[ln * NET_STRIDE + 2048 + r] =
            (__bf16)P.W[p][1 + j][(l * 16 + e) * 16 + lane];
    }
    // W5 (16x128): 8 N-tiles, compact (K = e, N = 16n + lane)
    for (int idx = tid; idx < 12 * 2048; idx += NTH) {
        int ln = idx >> 11, r = idx & 2047;
        int l = ln >> 1, p = ln & 1;
        int n = r >> 8, le = r & 255;
        int lane = le >> 4, e = le & 15;
        wfrag[ln * NET_STRIDE + 2816 + r] =
            (__bf16)P.W[p][4][(l * 16 + e) * 128 + 16 * n + lane];
    }
    // biases: per (layer,net) 192 floats: b1..b4 then b5[128]
    for (int idx = tid; idx < 12 * 192; idx += NTH) {
        int ln = idx / 192, r = idx - ln * 192;
        int l = ln >> 1, p = ln & 1;
        float v = (r < 64) ? P.b[p][r >> 4][l * 16 + (r & 15)]
                           : P.b[p][4][l * 128 + (r - 64)];
        bia[ln * 192 + r] = v;
    }
    for (int idx = tid; idx < 768; idx += NTH) msk[idx] = P.masks[idx];
    if (tid < 16) ((unsigned*)(smem + OFF_ZERO))[tid] = 0;   // zero page
    __syncthreads();

    const int wave   = tid >> 5;
    const int lane   = tid & 31;
    const int laneM  = lane & 15;   // row M in A layout, column N in C layout
    const int laneHi = lane >> 4;   // K-half / M-half selector
    const bool loK   = (lane < 16); // this lane holds K<16 of a B fragment

    float* hbuf = scratch + wave * 768;  // 16x16 f32 bounce
    float* tbuf = hbuf + 256;            // 16x16 f32 t head tile
    float* sbuf = tbuf + 256;            // 16x16 f32 s head tile

    const int tstride = gridDim.x * WPB;
    for (int tile = blockIdx.x * WPB + wave; tile < P.ntiles; tile += tstride) {
        const int row0 = tile * 16;
        const float* xrow = P.x + (size_t)(row0 + laneM) * 128 + 8 * laneHi;

        // x tile in registers, A-fragment-friendly layout (non-temporal read):
        // xr[8n+i] = x[row0+laneM][16n + 8*laneHi + i]
        float xr[64];
#pragma unroll
        for (int n = 0; n < 8; n++) {
            v4f lo = __builtin_nontemporal_load((const v4f*)(xrow + 16 * n));
            v4f hi = __builtin_nontemporal_load((const v4f*)(xrow + 16 * n + 4));
#pragma unroll
            for (int i = 0; i < 4; i++) {
                xr[8 * n + i]     = lo[i];
                xr[8 * n + 4 + i] = hi[i];
            }
        }
        // prefetch next tile (8KB, one cacheline-granular address per lane)
        if (tile + tstride < P.ntiles) {
            const float* nxt = P.x + (size_t)(tile + tstride) * 16 * 128 + lane * 64;
            __builtin_prefetch(nxt, 0, 1);
        }

        float ldacc = 0.0f;

        for (int it = 0; it < LAYERS; ++it) {
            const int l = LAYERS - 1 - it;            // layers in reverse
            const float* mrow = msk + l * 128;

            // masked input xm = x*mask as 4 bf16 A-fragments (K chunks of 32)
            v16bf A[4];
#pragma unroll
            for (int n = 0; n < 8; n++) {
                int c = n >> 1, eb = (n & 1) * 8;
                const float* mp = mrow + 16 * n + 8 * laneHi;
                v4f m0 = *(const v4f*)mp;
                v4f m1 = *(const v4f*)(mp + 4);
#pragma unroll
                for (int i = 0; i < 4; i++) {
                    A[c][eb + i]     = (__bf16)(xr[8 * n + i]     * m0[i]);
                    A[c][eb + 4 + i] = (__bf16)(xr[8 * n + 4 + i] * m1[i]);
                }
            }

            // run both MLP bodies (p=0:t, p=1:s) up to the final hidden layer
            v16bf Ah[2];
#pragma unroll
            for (int p = 0; p < 2; p++) {
                const __bf16* nb = wfrag + (l * 2 + p) * NET_STRIDE;
                const float*  bb = bia   + (l * 2 + p) * 192;

                v8f acc = {};                       // Dense(128->16)
#pragma unroll
                for (int c = 0; c < 4; c++) {
                    v16bf Bf = *(const v16bf*)(nb + c * 512 + lane * 16);
                    acc = __builtin_amdgcn_wmma_f32_16x16x32_bf16(
                        false, A[c], false, Bf, (short)0, acc, false, false);
                }
                v16bf h = c_to_afrag(acc, bb, hbuf, laneM, laneHi);

#pragma unroll
                for (int j = 0; j < 3; j++) {       // 3x Dense(16->16)
                    // lanes >=16 (K 16..31, zero-padded) read the LDS zero page:
                    // address cndmask instead of exec-masked load + zero fill
                    const __bf16* fp = loK ? (nb + 2048 + j * 256 + lane * 16)
                                           : zpage;
                    v16bf Bf = *(const v16bf*)fp;
                    v8f a2 = {};
                    a2 = __builtin_amdgcn_wmma_f32_16x16x32_bf16(
                        false, h, false, Bf, (short)0, a2, false, false);
                    h = c_to_afrag(a2, bb + 16 * (j + 1), hbuf, laneM, laneHi);
                }
                Ah[p] = h;
            }

            // heads: Dense(16->128) for t and s, fused with coupling update.
            const __bf16* nbt = wfrag + (l * 2 + 0) * NET_STRIDE + 2816;
            const __bf16* nbs = wfrag + (l * 2 + 1) * NET_STRIDE + 2816;
            const float*  b5t = bia + (l * 2 + 0) * 192 + 64;
            const float*  b5s = bia + (l * 2 + 1) * 192 + 64;
#pragma unroll 1
            for (int n = 0; n < 8; n++) {
                const __bf16* fpt = loK ? (nbt + n * 256 + lane * 16) : zpage;
                const __bf16* fps = loK ? (nbs + n * 256 + lane * 16) : zpage;
                v16bf Bt = *(const v16bf*)fpt;
                v16bf Bs = *(const v16bf*)fps;
                v8f ct = {};
                ct = __builtin_amdgcn_wmma_f32_16x16x32_bf16(
                    false, Ah[0], false, Bt, (short)0, ct, false, false);
                v8f cs = {};
                cs = __builtin_amdgcn_wmma_f32_16x16x32_bf16(
                    false, Ah[1], false, Bs, (short)0, cs, false, false);

                const float bt = b5t[16 * n + laneM];
                const float bs = b5s[16 * n + laneM];
#pragma unroll
                for (int r = 0; r < 8; r++) {
                    int M = r + 8 * laneHi;
                    tbuf[M * 16 + laneM] = ct[r] + bt;
                    sbuf[M * 16 + laneM] = cs[r] + bs;
                }
                const float* tp = tbuf + laneM * 16 + 8 * laneHi;
                const float* sp = sbuf + laneM * 16 + 8 * laneHi;
                const float* mp = mrow + 16 * n + 8 * laneHi;
                v4f m0 = *(const v4f*)mp;
                v4f m1 = *(const v4f*)(mp + 4);
                float mv[8] = {m0[0], m0[1], m0[2], m0[3],
                               m1[0], m1[1], m1[2], m1[3]};
#pragma unroll
                for (int i = 0; i < 8; i++) {
                    float rm = 1.0f - mv[i];
                    float sv = rm * fast_tanh(sp[i]);  // s = rm*tanh(net_s)
                    float xv = xr[8 * n + i];
                    xr[8 * n + i] = rm * (xv - tp[i]) * fast_exp(-sv) + mv[i] * xv;
                    ldacc -= sv;                       // logdet -= sum(s)
                }
            }
        }

        // write y tile (non-temporal: streamed once, keep L2 for weights)
        float* yrow = P.y + (size_t)(row0 + laneM) * 128 + 8 * laneHi;
#pragma unroll
        for (int n = 0; n < 8; n++) {
            v4f lo, hi;
#pragma unroll
            for (int i = 0; i < 4; i++) {
                lo[i] = xr[8 * n + i];
                hi[i] = xr[8 * n + 4 + i];
            }
            __builtin_nontemporal_store(lo, (v4f*)(yrow + 16 * n));
            __builtin_nontemporal_store(hi, (v4f*)(yrow + 16 * n + 4));
        }
        float other = __shfl_xor(ldacc, 16, 32);  // combine the two half-row lanes
        if (lane < 16)
            __builtin_nontemporal_store(ldacc + other, P.ld + row0 + laneM);
    }
}

extern "C" void kernel_launch(void* const* d_in, const int* in_sizes, int n_in,
                              void* d_out, int out_size, void* d_ws, size_t ws_size,
                              hipStream_t stream) {
    (void)n_in; (void)out_size; (void)d_ws; (void)ws_size;

    Params P;
    P.x     = (const float*)d_in[0];
    P.masks = (const float*)d_in[1];
    for (int p = 0; p < 2; p++)
        for (int j = 0; j < 5; j++) {
            P.W[p][j] = (const float*)d_in[2 + p * 10 + j * 2];
            P.b[p][j] = (const float*)d_in[2 + p * 10 + j * 2 + 1];
        }
    const int rows = in_sizes[0] / 128;
    P.y      = (float*)d_out;
    P.ld     = (float*)d_out + (size_t)rows * 128;
    P.ntiles = rows / 16;

    (void)hipFuncSetAttribute(reinterpret_cast<const void*>(realnvp_fused),
                              hipFuncAttributeMaxDynamicSharedMemorySize,
                              (int)SMEM_BYTES);

    int blocks = 512;                                   // 16 tiles per block
    if (blocks * WPB > P.ntiles) blocks = (P.ntiles + WPB - 1) / WPB;
    realnvp_fused<<<blocks, NTH, SMEM_BYTES, stream>>>(P);
}
